// RG_6708738917104
// MI455X (gfx1250) — matrix operand order repaired and checked by reading, not compiled, not run on hardware
//
#include <hip/hip_runtime.h>
#include <hip/hip_bf16.h>

// ---------------------------------------------------------------------------
// CDNA5 (gfx1250) fused "recurrent gate" kernel set.
//   out = exp(c[d]*sigmoid(x@Wa + ba)) * pooled[b,d]
//       + x * sigmoid(x@Wi + bi) * sqrt(1 - exp(2*c[d]*sigmoid(x@Wa + ba)))
// Block-diagonal GEMMs run on v_wmma_f32_16x16x32_f16 (f16 in, f32 acc).
// 64-row M tiles: B fragments loaded once per K-step and reused over 4 M
// sub-tiles; all 4 A fragments prefetched per K-step so the 8 WMMAs issue
// back-to-back behind a single s_wait_dscnt.
// ---------------------------------------------------------------------------

typedef __attribute__((ext_vector_type(16))) _Float16 v16h;
typedef __attribute__((ext_vector_type(8)))  _Float16 v8h;
typedef __attribute__((ext_vector_type(8)))  float    v8f;

#define MODEL_D 2048
#define DH      256
#define HEADS   8
#define MROWS   64          // rows (M) per block tile
#define MSUB    (MROWS/16)  // 4 M sub-tiles of 16 rows
#define XPAD    8           // LDS row pad (halves) -> conflict-free ds_load_b128
#define NT_POOL 64          // T chunks for deterministic two-stage pooling

// ---------------------------------------------------------------------------
// K0: weight transpose + f32->f16 convert:  wT[h][j][i] = (f16) w[h][i][j]
// ---------------------------------------------------------------------------
__global__ __launch_bounds__(256) void rg_prep_weights(
    const float* __restrict__ w, _Float16* __restrict__ wT, int total) {
  for (int i = blockIdx.x * blockDim.x + threadIdx.x; i < total;
       i += gridDim.x * blockDim.x) {
    int hd  = i / (DH * DH);
    int rem = i - hd * (DH * DH);
    int j   = rem / DH;       // output column
    int ii  = rem - j * DH;   // input (K) index
    wT[((size_t)hd * DH + j) * DH + ii] =
        (_Float16)w[((size_t)hd * DH + ii) * DH + j];
  }
}

// ---------------------------------------------------------------------------
// K1: c[d] = -8 * softplus(a_param[d])   (stable softplus)
// ---------------------------------------------------------------------------
__global__ __launch_bounds__(256) void rg_prep_cvec(
    const float* __restrict__ ap, float* __restrict__ c, int n) {
  int i = blockIdx.x * blockDim.x + threadIdx.x;
  if (i < n) {
    float v  = ap[i];
    float sp = (v > 20.f) ? v : log1pf(__expf(v));
    c[i] = -8.f * sp;
  }
}

// ---------------------------------------------------------------------------
// K2: logits[b,t,h] = mean over dh of h   (one block per (b,t) row)
// ---------------------------------------------------------------------------
__global__ __launch_bounds__(256) void rg_pool_logits(
    const float* __restrict__ hin, float* __restrict__ logits) {
  __shared__ float red[256];
  const size_t bt = blockIdx.x;
  const float* row = hin + bt * MODEL_D;
  float s = 0.f;
  const int d0 = threadIdx.x * 8;
#pragma unroll
  for (int k = 0; k < 8; ++k) s += row[d0 + k];
  red[threadIdx.x] = s;
  __syncthreads();
  if (threadIdx.x < HEADS) {
    float t = 0.f;
#pragma unroll
    for (int k = 0; k < 32; ++k) t += red[threadIdx.x * 32 + k];
    logits[bt * HEADS + threadIdx.x] = t * (1.0f / DH);
  }
}

// ---------------------------------------------------------------------------
// K3: per-(b,h) softmax stats over T: row max and sum of exp
// ---------------------------------------------------------------------------
__global__ __launch_bounds__(256) void rg_softmax_stats(
    const float* __restrict__ logits, float* __restrict__ smax,
    float* __restrict__ ssum, int T) {
  __shared__ float red[256];
  const int bh = blockIdx.x;
  const int b  = bh / HEADS, hh = bh % HEADS;
  const float* lp = logits + ((size_t)b * T) * HEADS + hh;

  float m = -1e30f;
  for (int t = threadIdx.x; t < T; t += 256)
    m = fmaxf(m, lp[(size_t)t * HEADS]);
  red[threadIdx.x] = m;
  __syncthreads();
  for (int off = 128; off > 0; off >>= 1) {
    if (threadIdx.x < off)
      red[threadIdx.x] = fmaxf(red[threadIdx.x], red[threadIdx.x + off]);
    __syncthreads();
  }
  m = red[0];
  __syncthreads();

  float s = 0.f;
  for (int t = threadIdx.x; t < T; t += 256)
    s += __expf(lp[(size_t)t * HEADS] - m);
  red[threadIdx.x] = s;
  __syncthreads();
  for (int off = 128; off > 0; off >>= 1) {
    if (threadIdx.x < off) red[threadIdx.x] += red[threadIdx.x + off];
    __syncthreads();
  }
  if (threadIdx.x == 0) { smax[bh] = m; ssum[bh] = red[0]; }
}

// ---------------------------------------------------------------------------
// K4: partial weighted pooling over a T-chunk (deterministic, no atomics)
// ---------------------------------------------------------------------------
__global__ __launch_bounds__(256) void rg_pool_partial(
    const float* __restrict__ hin, const float* __restrict__ logits,
    const float* __restrict__ smax, float* __restrict__ partial, int T) {
  const int blk = blockIdx.x;
  const int bh  = blk / NT_POOL;
  const int ch  = blk % NT_POOL;
  const int b   = bh / HEADS, hh = bh % HEADS;
  const int tch = T / NT_POOL;
  const int t0  = ch * tch;
  const float mx = smax[bh];
  float acc = 0.f;
  for (int tt = 0; tt < tch; ++tt) {
    const size_t bt = (size_t)b * T + (t0 + tt);
    const float w = __expf(logits[bt * HEADS + hh] - mx);
    acc += w * hin[bt * MODEL_D + hh * DH + threadIdx.x];
  }
  partial[((size_t)bh * NT_POOL + ch) * DH + threadIdx.x] = acc;
}

// K5: reduce partials and normalize: pooled[b, h*DH + dh]
__global__ __launch_bounds__(256) void rg_pool_reduce(
    const float* __restrict__ partial, const float* __restrict__ ssum,
    float* __restrict__ pooled) {
  const int bh = blockIdx.x;
  const int b  = bh / HEADS, hh = bh % HEADS;
  float s = 0.f;
  for (int ch = 0; ch < NT_POOL; ++ch)
    s += partial[((size_t)bh * NT_POOL + ch) * DH + threadIdx.x];
  pooled[(size_t)b * MODEL_D + hh * DH + threadIdx.x] = s / ssum[bh];
}

// ---------------------------------------------------------------------------
// K6: main fused kernel. 64-row M tile per block, 256 threads = 8 waves.
// Per head: stage x[64, dh] -> LDS f16. Each wave owns two 16-wide N tiles;
// per K-step: load both B fragments (global, L2-resident) and ALL FOUR A
// fragments (LDS) first, then fire 8 WMMAs back-to-back. Fused epilogue.
// ---------------------------------------------------------------------------
__global__ __launch_bounds__(256) void rg_fused_gate(
    const float* __restrict__ x, const float* __restrict__ bin,
    const float* __restrict__ ba, const _Float16* __restrict__ wTx,
    const _Float16* __restrict__ wTa, const float* __restrict__ cvec,
    const float* __restrict__ pooled, float* __restrict__ out, int T) {
  __shared__ __align__(16) _Float16 xs[MROWS][DH + XPAD];

  const int tid     = threadIdx.x;
  const int lane    = tid & 31;          // wave32
  const int wave    = tid >> 5;          // 8 waves
  const int rowBase = blockIdx.x * MROWS;       // flat over B*T
  const int b       = rowBase / T;              // whole tile in one batch

  for (int hd = 0; hd < HEADS; ++hd) {
    // ---- stage this head's x slice into LDS as f16 (64 rows x 256 cols),
    //      coalesced: consecutive lanes load consecutive float4s.
    {
      const int c0 = (tid & 63) * 4;     // 0..252, float4 column
      const int r0 = tid >> 6;           // 0..3
#pragma unroll
      for (int pass = 0; pass < 16; ++pass) {
        const int r = r0 + pass * 4;
        const float4 v = *(const float4*)(
            x + (size_t)(rowBase + r) * MODEL_D + hd * DH + c0);
        _Float16* d = &xs[r][c0];
        d[0] = (_Float16)v.x; d[1] = (_Float16)v.y;
        d[2] = (_Float16)v.z; d[3] = (_Float16)v.w;
      }
    }
    __syncthreads();

    // ---- A-fragment lane mapping (ISA 16-bit A 16x32 layout):
    //   lanes 0-15 : row = lane,    halves = K[0..7]  ++ K[16..23]
    //   lanes 16-31: row = lane-16, halves = K[8..15] ++ K[24..31]
    const int arow  = lane & 15;
    const int koff  = (lane >> 4) * 8;   // A half-chunk offset
    const int bkoff = (lane >> 4) * 8;   // B: lanes 0-15 -> K0..7,16..23; 16-31 -> K8..15,24..31

#pragma unroll
    for (int nt = 0; nt < 2; ++nt) {
      const int ntile     = wave * 2 + nt;           // 0..15 within head
      const int colInHead = ntile * 16 + (lane & 15);
      const int dcol      = hd * DH + colInHead;
      const _Float16* wxp = wTx + (size_t)dcol * DH; // column of W (K-contig)
      const _Float16* wap = wTa + (size_t)dcol * DH;

      v8f accX[MSUB];
      v8f accA[MSUB];
      const v8f vzero = {};
#pragma unroll
      for (int m = 0; m < MSUB; ++m) { accX[m] = vzero; accA[m] = vzero; }

#pragma unroll
      for (int kb = 0; kb < 8; ++kb) {
        // B fragments for this K step (loaded once, reused over MSUB)
        union frag { v16h v; v8h h[2]; };
        frag bx, bav;
        const _Float16* bp = wxp + kb * 32 + bkoff;
        bx.h[0] = *(const v8h*)bp;
        bx.h[1] = *(const v8h*)(bp + 16);
        const _Float16* bp2 = wap + kb * 32 + bkoff;
        bav.h[0] = *(const v8h*)bp2;
        bav.h[1] = *(const v8h*)(bp2 + 16);

        // prefetch ALL A fragments for this K step (8x ds_load_b128,
        // one wait, then 8 back-to-back WMMAs)
        frag a[MSUB];
#pragma unroll
        for (int m = 0; m < MSUB; ++m) {
          const _Float16* apx = &xs[m * 16 + arow][kb * 32 + koff];
          a[m].h[0] = *(const v8h*)apx;             // K block low
          a[m].h[1] = *(const v8h*)(apx + 16);      // K block high (+16 halves)
        }
#pragma unroll
        for (int m = 0; m < MSUB; ++m) {
          accX[m] = __builtin_amdgcn_wmma_f32_16x16x32_f16(
              false, a[m].v, false, bx.v, (short)0, accX[m], false, false);
          accA[m] = __builtin_amdgcn_wmma_f32_16x16x32_f16(
              false, a[m].v, false, bav.v, (short)0, accA[m], false, false);
        }
      }

      // ---- fused epilogue. C/D layout: lane n=lane&15, rows i + 8*(lane>=16)
      const float bi   = bin[hd * DH + colInHead];
      const float bva  = ba[hd * DH + colInHead];
      const float c    = cvec[dcol];
      const float hp   = pooled[(size_t)b * MODEL_D + dcol];
      const int   roff = (lane >> 4) * 8;
#pragma unroll
      for (int m = 0; m < MSUB; ++m) {
#pragma unroll
        for (int i = 0; i < 8; ++i) {
          const int grow = rowBase + m * 16 + roff + i;
          const float gx = 1.f / (1.f + __expf(-(accX[m][i] + bi)));
          const float ga = 1.f / (1.f + __expf(-(accA[m][i] + bva)));
          const float la = c * ga;              // log_a <= 0
          const float av = __expf(la);
          const float sc = sqrtf(fmaxf(1.f - __expf(2.f * la), 0.f));
          const float xv = x[(size_t)grow * MODEL_D + dcol];
          out[(size_t)grow * MODEL_D + dcol] = av * hp + xv * gx * sc;
        }
      }
    }
    __syncthreads();
  }
}

// ---------------------------------------------------------------------------
// Host launcher
// ---------------------------------------------------------------------------
extern "C" void kernel_launch(void* const* d_in, const int* in_sizes, int n_in,
                              void* d_out, int out_size, void* d_ws, size_t ws_size,
                              hipStream_t stream) {
  const float* x       = (const float*)d_in[0];
  const float* h       = (const float*)d_in[1];
  const float* w_input = (const float*)d_in[2];
  const float* b_input = (const float*)d_in[3];
  const float* w_a     = (const float*)d_in[4];
  const float* b_a     = (const float*)d_in[5];
  const float* a_param = (const float*)d_in[6];
  float* out = (float*)d_out;

  const int  T  = 8192;
  const long long BT = (long long)in_sizes[0] / MODEL_D;  // B*T
  const int  B  = (int)(BT / T);

  // workspace carve-up (256-byte aligned slabs)
  char* p = (char*)d_ws;
  auto carve = [&](size_t bytes) {
    char* r = p;
    p += (bytes + 255) & ~(size_t)255;
    return r;
  };
  const size_t wElems = (size_t)HEADS * DH * DH;
  _Float16* wTx    = (_Float16*)carve(sizeof(_Float16) * wElems);
  _Float16* wTa    = (_Float16*)carve(sizeof(_Float16) * wElems);
  float*    cvec   = (float*)carve(sizeof(float) * MODEL_D);
  float*    logits = (float*)carve(sizeof(float) * (size_t)BT * HEADS);
  float*    smax   = (float*)carve(sizeof(float) * B * HEADS);
  float*    ssum   = (float*)carve(sizeof(float) * B * HEADS);
  float*    part   = (float*)carve(sizeof(float) * (size_t)B * HEADS * NT_POOL * DH);
  float*    pooled = (float*)carve(sizeof(float) * (size_t)B * MODEL_D);

  // K0: weight convert/transpose (both matrices)
  rg_prep_weights<<<2048, 256, 0, stream>>>(w_input, wTx, (int)wElems);
  rg_prep_weights<<<2048, 256, 0, stream>>>(w_a, wTa, (int)wElems);
  // K1: per-channel decay constant
  rg_prep_cvec<<<(MODEL_D + 255) / 256, 256, 0, stream>>>(a_param, cvec, MODEL_D);
  // K2: head logits
  rg_pool_logits<<<(int)BT, 256, 0, stream>>>(h, logits);
  // K3: softmax stats over T
  rg_softmax_stats<<<B * HEADS, 256, 0, stream>>>(logits, smax, ssum, T);
  // K4/K5: deterministic two-stage attention pooling
  rg_pool_partial<<<B * HEADS * NT_POOL, 256, 0, stream>>>(h, logits, smax, part, T);
  rg_pool_reduce<<<B * HEADS, 256, 0, stream>>>(part, ssum, pooled);
  // K6: fused WMMA gate kernel
  rg_fused_gate<<<(int)(BT / MROWS), 256, 0, stream>>>(
      x, b_input, b_a, wTx, wTa, cvec, pooled, out, T);
}
